// Attention_54692113547630
// MI455X (gfx1250) — matrix operand order repaired and checked by reading, not compiled, not run on hardware
//
#include <hip/hip_runtime.h>
#include <hip/hip_bf16.h>

// ---------------------------------------------------------------------------
// CDNA5 (gfx1250) fused attention block:
//   LN -> QKV GEMM (bf16 WMMA) -> flash attention (bf16 WMMA, online softmax)
//   -> out GEMM (bf16 WMMA, f32 out + bias)
// Matmuls: v_wmma_f32_16x16x32_bf16 (wave32).
// Data movement: global_load_async_to_lds_b128 + s_wait_asynccnt,
// double-buffered LDS tiles. Softmax row-max via ds_swizzle XOR butterflies;
// softmax row-sum via WMMA (P @ ones).
// ---------------------------------------------------------------------------

typedef __attribute__((ext_vector_type(16))) __bf16 bf16x16;
typedef __attribute__((ext_vector_type(8)))  float  f32x8;

#define D_MODEL 1024
#define D3      3072
#define SEQ     2048
#define BATCH   2
#define HEADS   16
#define DHEAD   64
#define ROWS    (BATCH * SEQ)      // 4096
#define ATTN_SCALE 0.125f          // 64^-0.5

__device__ __forceinline__ unsigned short f2bf(float f) {
    unsigned int u = __float_as_uint(f);
    u += 0x7FFFu + ((u >> 16) & 1u);   // round-to-nearest-even
    return (unsigned short)(u >> 16);
}

// Build a 16-element bf16 fragment from two 16-byte chunks (8 bf16 each).
__device__ __forceinline__ bf16x16 load_frag(const unsigned short* p0,
                                             const unsigned short* p1) {
    union { bf16x16 v; uint4 u[2]; } f;
    f.u[0] = *reinterpret_cast<const uint4*>(p0);
    f.u[1] = *reinterpret_cast<const uint4*>(p1);
    return f.v;
}

__device__ __forceinline__ bf16x16 ones_frag() {
    union { bf16x16 v; unsigned int u[8]; } f;
    #pragma unroll
    for (int i = 0; i < 8; ++i) f.u[i] = 0x3F803F80u;   // bf16 1.0 pairs
    return f.v;
}

__device__ __forceinline__ f32x8 wmma_bf16(bf16x16 a, bf16x16 b, f32x8 c) {
    return __builtin_amdgcn_wmma_f32_16x16x32_bf16(
        /*neg_a=*/false, a, /*neg_b=*/false, b,
        /*c_mod=*/(short)0, c, /*reuse_a=*/false, /*reuse_b=*/false);
}

// Async 16B copy global->LDS (per-lane), tracked by ASYNCcnt.
__device__ __forceinline__ void async_copy_b128(const unsigned short* g,
                                                unsigned short* l) {
    unsigned loff = (unsigned)(unsigned long long)l;   // low 32b = LDS offset
    asm volatile("global_load_async_to_lds_b128 %0, %1, off"
                 :: "v"(loff), "v"((unsigned long long)g)
                 : "memory");
}
__device__ __forceinline__ void wait_async0() {
    asm volatile("s_wait_asynccnt 0" ::: "memory");
}

// max across the 16-lane row group via ds_swizzle XOR butterflies
// (group-of-32 mode: imm = xor_mask<<10 | and_mask 0x1F)
__device__ __forceinline__ float rowmax16(float x) {
    x = fmaxf(x, __uint_as_float(
            __builtin_amdgcn_ds_swizzle(__float_as_uint(x), 0x041F)));
    x = fmaxf(x, __uint_as_float(
            __builtin_amdgcn_ds_swizzle(__float_as_uint(x), 0x081F)));
    x = fmaxf(x, __uint_as_float(
            __builtin_amdgcn_ds_swizzle(__float_as_uint(x), 0x101F)));
    x = fmaxf(x, __uint_as_float(
            __builtin_amdgcn_ds_swizzle(__float_as_uint(x), 0x201F)));
    return x;
}

// ---------------------------------------------------------------------------
// Transpose + f32->bf16 cast of a weight matrix: in [K][N] -> outT [N][K]
// ---------------------------------------------------------------------------
__global__ void transpose_cast_kernel(const float* __restrict__ in,
                                      unsigned short* __restrict__ outT,
                                      int K, int N) {
    __shared__ unsigned short t[64][72];
    const int k0 = blockIdx.y * 64;
    const int n0 = blockIdx.x * 64;
    const int tid = threadIdx.x;
    #pragma unroll
    for (int i = 0; i < 4; ++i) {
        int chunk = tid + i * 256;
        int r = chunk >> 4;                // k row 0..63
        int c = (chunk & 15) * 4;          // n col
        float4 v = *reinterpret_cast<const float4*>(
            in + (size_t)(k0 + r) * N + n0 + c);
        t[c + 0][r] = f2bf(v.x);
        t[c + 1][r] = f2bf(v.y);
        t[c + 2][r] = f2bf(v.z);
        t[c + 3][r] = f2bf(v.w);
    }
    __syncthreads();
    #pragma unroll
    for (int i = 0; i < 2; ++i) {
        int chunk = tid + i * 256;
        int r = chunk >> 3;                // n row 0..63
        int c = (chunk & 7) * 8;           // k col
        uint4 d = *reinterpret_cast<const uint4*>(&t[r][c]);
        *reinterpret_cast<uint4*>(outT + (size_t)(n0 + r) * K + k0 + c) = d;
    }
}

// ---------------------------------------------------------------------------
// LayerNorm over D=1024, one row per 256-thread block, bf16 output
// ---------------------------------------------------------------------------
__global__ void ln_cast_kernel(const float* __restrict__ x,
                               const float* __restrict__ gamma,
                               const float* __restrict__ beta,
                               unsigned short* __restrict__ xn) {
    const int row = blockIdx.x;
    const int t = threadIdx.x;
    __shared__ float red[256];
    const float* xr = x + (size_t)row * D_MODEL;
    float4 v = reinterpret_cast<const float4*>(xr)[t];
    red[t] = v.x + v.y + v.z + v.w;
    __syncthreads();
    #pragma unroll
    for (int o = 128; o > 0; o >>= 1) {
        if (t < o) red[t] += red[t + o];
        __syncthreads();
    }
    float mean = red[0] * (1.0f / D_MODEL);
    __syncthreads();
    float cx = v.x - mean, cy = v.y - mean, cz = v.z - mean, cw = v.w - mean;
    red[t] = cx * cx + cy * cy + cz * cz + cw * cw;
    __syncthreads();
    #pragma unroll
    for (int o = 128; o > 0; o >>= 1) {
        if (t < o) red[t] += red[t + o];
        __syncthreads();
    }
    float r = rsqrtf(red[0] * (1.0f / D_MODEL) + 1e-5f);
    float4 g  = reinterpret_cast<const float4*>(gamma)[t];
    float4 bb = reinterpret_cast<const float4*>(beta)[t];
    float o0 = cx * r * g.x + bb.x;
    float o1 = cy * r * g.y + bb.y;
    float o2 = cz * r * g.z + bb.z;
    float o3 = cw * r * g.w + bb.w;
    unsigned int lo = (unsigned int)f2bf(o0) | ((unsigned int)f2bf(o1) << 16);
    unsigned int hi = (unsigned int)f2bf(o2) | ((unsigned int)f2bf(o3) << 16);
    reinterpret_cast<uint2*>(xn + (size_t)row * D_MODEL)[t] = make_uint2(lo, hi);
}

// ---------------------------------------------------------------------------
// bf16 GEMM: C[MxN] = A[MxK] @ Bt^T  (Bt pre-transposed: [N][K]).
// 128x128 block tile, BK=32, 8 waves, wave tile 32x64 (2x4 WMMA tiles).
// Tiles stream via async-to-LDS, double-buffered.
// MODE 0: bf16 out; Q cols pre-scaled; V cols routed to transposed vt buffer.
// MODE 1: f32 out + bias.
// ---------------------------------------------------------------------------
template <int MODE>
__global__ void gemm_bf16_kernel(const unsigned short* __restrict__ A,
                                 const unsigned short* __restrict__ Bt,
                                 unsigned short* __restrict__ Cbf,
                                 unsigned short* __restrict__ Vbf,
                                 float* __restrict__ Cf,
                                 const float* __restrict__ bias,
                                 int M, int N, int K) {
    __shared__ unsigned short Al[2][128 * 40];
    __shared__ unsigned short Bl[2][128 * 40];
    const int tid  = threadIdx.x;
    const int lane = tid & 31;
    const int wave = tid >> 5;
    const int wm   = wave >> 1;
    const int wn   = wave & 1;
    const int lrow = lane & 15;
    const int lhi  = lane >> 4;
    const int koff = lhi * 16;
    const int m0 = blockIdx.y * 128;
    const int n0 = blockIdx.x * 128;
    const int cr = tid >> 2;
    const int cc = (tid & 3) * 8;

    f32x8 zero = {};
    f32x8 acc[2][4];
    #pragma unroll
    for (int mt = 0; mt < 2; ++mt)
        #pragma unroll
        for (int nt = 0; nt < 4; ++nt) acc[mt][nt] = zero;

    auto issue = [&](int buf, int k0) {
        #pragma unroll
        for (int i = 0; i < 2; ++i) {
            int r = cr + i * 64;
            async_copy_b128(A  + (size_t)(m0 + r) * K + k0 + cc,
                            &Al[buf][r * 40 + cc]);
            async_copy_b128(Bt + (size_t)(n0 + r) * K + k0 + cc,
                            &Bl[buf][r * 40 + cc]);
        }
    };

    issue(0, 0);
    int cur = 0;
    for (int k0 = 0; k0 < K; k0 += 32) {
        wait_async0();
        __syncthreads();
        if (k0 + 32 < K) issue(cur ^ 1, k0 + 32);

        bf16x16 af[2], bfr[4];
        #pragma unroll
        for (int mt = 0; mt < 2; ++mt) {
            const unsigned short* p = &Al[cur][(wm * 32 + mt * 16 + lrow) * 40];
            af[mt] = load_frag(p + koff, p + koff + 8);
        }
        #pragma unroll
        for (int nt = 0; nt < 4; ++nt) {
            const unsigned short* p =
                &Bl[cur][(wn * 64 + nt * 16 + lrow) * 40 + lhi * 16];
            bfr[nt] = load_frag(p, p + 8);
        }
        #pragma unroll
        for (int mt = 0; mt < 2; ++mt)
            #pragma unroll
            for (int nt = 0; nt < 4; ++nt)
                acc[mt][nt] = wmma_bf16(af[mt], bfr[nt], acc[mt][nt]);
        cur ^= 1;
    }

    // epilogue: region test is uniform per 128-wide tile
    const int rbase = m0 + wm * 32 + lhi * 8;
    const int cbase = n0 + wn * 64 + lrow;
    if (MODE == 1) {
        #pragma unroll
        for (int nt = 0; nt < 4; ++nt) {
            int col = cbase + nt * 16;
            float bv = bias[col];
            #pragma unroll
            for (int mt = 0; mt < 2; ++mt)
                #pragma unroll
                for (int v = 0; v < 8; ++v)
                    Cf[(size_t)(rbase + mt * 16 + v) * N + col] =
                        acc[mt][nt][v] + bv;
        }
    } else if (n0 < D_MODEL) {                       // Q region: pre-scale
        #pragma unroll
        for (int mt = 0; mt < 2; ++mt)
            #pragma unroll
            for (int nt = 0; nt < 4; ++nt)
                #pragma unroll
                for (int v = 0; v < 8; ++v)
                    Cbf[(size_t)(rbase + mt * 16 + v) * N + cbase + nt * 16] =
                        f2bf(acc[mt][nt][v] * ATTN_SCALE);
    } else if (n0 < 2 * D_MODEL) {                   // K region
        #pragma unroll
        for (int mt = 0; mt < 2; ++mt)
            #pragma unroll
            for (int nt = 0; nt < 4; ++nt)
                #pragma unroll
                for (int v = 0; v < 8; ++v)
                    Cbf[(size_t)(rbase + mt * 16 + v) * N + cbase + nt * 16] =
                        f2bf(acc[mt][nt][v]);
    } else {                                         // V region -> vt[b*D+d][s]
        #pragma unroll
        for (int mt = 0; mt < 2; ++mt)
            #pragma unroll
            for (int nt = 0; nt < 4; ++nt)
                #pragma unroll
                for (int v = 0; v < 8; ++v) {
                    int row = rbase + mt * 16 + v;
                    int d   = cbase + nt * 16 - 2 * D_MODEL;
                    Vbf[((size_t)((row >> 11) * D_MODEL + d)) * SEQ +
                        (row & (SEQ - 1))] = f2bf(acc[mt][nt][v]);
                }
    }
}

// ---------------------------------------------------------------------------
// Flash attention: 8 waves (256 thr) per block, 128 queries/block,
// 16 queries per wave. K/V tiles stream async double-buffered; row-max via
// ds_swizzle; row-sum via WMMA (P @ ones).
// ---------------------------------------------------------------------------
__global__ void attn_kernel(const unsigned short* __restrict__ qkv,
                            const unsigned short* __restrict__ vt,
                            unsigned short* __restrict__ attn) {
    const int qt = blockIdx.x;     // query tile 0..15 (128 queries each)
    const int h  = blockIdx.y;
    const int b  = blockIdx.z;
    const int tid  = threadIdx.x;  // 0..255
    const int lane = tid & 31;
    const int wave = tid >> 5;     // 0..7
    const int lrow = lane & 15;
    const int lhi  = lane >> 4;

    __shared__ unsigned short Kl[2][64 * 72];   // [key][dh]
    __shared__ unsigned short Vl[2][64 * 72];   // [dh][key]
    __shared__ unsigned short Pl[8][16 * 72];   // per-wave P staging

    const unsigned short* qp = qkv
        + ((size_t)(b * SEQ + qt * 128 + wave * 16 + lrow)) * D3 + h * DHEAD;
    bf16x16 aq[2];
    #pragma unroll
    for (int ks = 0; ks < 2; ++ks) {
        const unsigned short* p = qp + ks * 32 + lhi * 16;
        aq[ks] = load_frag(p, p + 8);
    }
    const bf16x16 ones = ones_frag();

    const unsigned short* Kg0 = qkv + D_MODEL + h * DHEAD
                              + (size_t)(b * SEQ) * D3;
    const unsigned short* Vg0 = vt + (size_t)(b * D_MODEL + h * DHEAD) * SEQ;
    const int cr = tid >> 3;           // 0..31 (+32)
    const int cc = (tid & 7) * 8;

    auto issue = [&](int buf, int kt) {
        #pragma unroll
        for (int i = 0; i < 2; ++i) {
            int r = cr + i * 32;       // 0..63
            async_copy_b128(Kg0 + (size_t)(kt * 64 + r) * D3 + cc,
                            &Kl[buf][r * 72 + cc]);
            async_copy_b128(Vg0 + (size_t)r * SEQ + kt * 64 + cc,
                            &Vl[buf][r * 72 + cc]);
        }
    };

    f32x8 zero = {};
    f32x8 acc[4];
    #pragma unroll
    for (int nt = 0; nt < 4; ++nt) acc[nt] = zero;
    float mrow[8], lsum[8], alph[8];
    #pragma unroll
    for (int v = 0; v < 8; ++v) { mrow[v] = -1e30f; lsum[v] = 0.0f; }

    issue(0, 0);
    int cur = 0;
    for (int kt = 0; kt < SEQ / 64; ++kt) {
        wait_async0();
        __syncthreads();
        if (kt + 1 < SEQ / 64) issue(cur ^ 1, kt + 1);

        // S = Q @ K^T : 4 N-tiles of 16 keys, dh=64 -> 2 k-steps
        f32x8 s[4];
        #pragma unroll
        for (int nt = 0; nt < 4; ++nt) {
            f32x8 c0 = zero;
            #pragma unroll
            for (int ks = 0; ks < 2; ++ks) {
                const unsigned short* p =
                    &Kl[cur][(nt * 16 + lrow) * 72 + ks * 32 + lhi * 16];
                bf16x16 bk = load_frag(p, p + 8);
                c0 = wmma_bf16(aq[ks], bk, c0);
            }
            s[nt] = c0;
        }

        // online softmax: max via ds_swizzle butterflies
        #pragma unroll
        for (int v = 0; v < 8; ++v) {
            float mx = fmaxf(fmaxf(s[0][v], s[1][v]), fmaxf(s[2][v], s[3][v]));
            mx = rowmax16(mx);
            float mnew = fmaxf(mrow[v], mx);
            alph[v] = __expf(mrow[v] - mnew);
            mrow[v] = mnew;
            #pragma unroll
            for (int nt = 0; nt < 4; ++nt)
                s[nt][v] = __expf(s[nt][v] - mnew);
            #pragma unroll
            for (int nt = 0; nt < 4; ++nt) acc[nt][v] *= alph[v];
        }

        // stage P (C-layout -> LDS row-major) for A-fragment reload
        #pragma unroll
        for (int nt = 0; nt < 4; ++nt)
            #pragma unroll
            for (int v = 0; v < 8; ++v)
                Pl[wave][(lhi * 8 + v) * 72 + nt * 16 + lrow] = f2bf(s[nt][v]);

        // reload P as A-fragments; row-sum via WMMA (P @ ones); then P @ V
        bf16x16 ap[2];
        #pragma unroll
        for (int ks = 0; ks < 2; ++ks) {
            const unsigned short* pp = &Pl[wave][lrow * 72 + ks * 32 + lhi * 16];
            ap[ks] = load_frag(pp, pp + 8);
        }
        f32x8 rs = zero;
        rs = wmma_bf16(ap[0], ones, rs);
        rs = wmma_bf16(ap[1], ones, rs);
        #pragma unroll
        for (int v = 0; v < 8; ++v) lsum[v] = lsum[v] * alph[v] + rs[v];

        #pragma unroll
        for (int ks = 0; ks < 2; ++ks)
            #pragma unroll
            for (int nt = 0; nt < 4; ++nt) {
                const unsigned short* pv =
                    &Vl[cur][(nt * 16 + lrow) * 72 + ks * 32 + lhi * 16];
                bf16x16 bv = load_frag(pv, pv + 8);
                acc[nt] = wmma_bf16(ap[ks], bv, acc[nt]);
            }
        cur ^= 1;
    }

    // epilogue: O /= l, write bf16 head-merged [B*S, D]
    #pragma unroll
    for (int nt = 0; nt < 4; ++nt) {
        #pragma unroll
        for (int v = 0; v < 8; ++v) {
            int r = qt * 128 + wave * 16 + lhi * 8 + v;
            int c = h * DHEAD + nt * 16 + lrow;
            attn[((size_t)(b * SEQ + r)) * D_MODEL + c] =
                f2bf(acc[nt][v] / lsum[v]);
        }
    }
}

// ---------------------------------------------------------------------------
extern "C" void kernel_launch(void* const* d_in, const int* in_sizes, int n_in,
                              void* d_out, int out_size, void* d_ws, size_t ws_size,
                              hipStream_t stream) {
    const float* x     = (const float*)d_in[0];
    const float* gamma = (const float*)d_in[1];
    const float* beta  = (const float*)d_in[2];
    const float* w_qkv = (const float*)d_in[3];
    const float* w_out = (const float*)d_in[4];
    const float* b_out = (const float*)d_in[5];
    float* out = (float*)d_out;

    unsigned short* xn   = (unsigned short*)d_ws;                 // 4096*1024
    unsigned short* wqT  = xn   + (size_t)ROWS * D_MODEL;         // [3072][1024]
    unsigned short* woT  = wqT  + (size_t)D_MODEL * D3;           // [1024][1024]
    unsigned short* qkv  = woT  + (size_t)D_MODEL * D_MODEL;      // 4096*3072
    unsigned short* attn = qkv  + (size_t)ROWS * D3;              // 4096*1024
    unsigned short* vt   = attn + (size_t)ROWS * D_MODEL;         // [2*1024][2048]

    transpose_cast_kernel<<<dim3(D3 / 64, D_MODEL / 64), 256, 0, stream>>>(
        w_qkv, wqT, D_MODEL, D3);
    transpose_cast_kernel<<<dim3(D_MODEL / 64, D_MODEL / 64), 256, 0, stream>>>(
        w_out, woT, D_MODEL, D_MODEL);

    ln_cast_kernel<<<ROWS, 256, 0, stream>>>(x, gamma, beta, xn);

    gemm_bf16_kernel<0><<<dim3(D3 / 128, ROWS / 128), 256, 0, stream>>>(
        xn, wqT, qkv, vt, nullptr, nullptr, ROWS, D3, D_MODEL);

    attn_kernel<<<dim3(SEQ / 128, HEADS, BATCH), 256, 0, stream>>>(qkv, vt, attn);

    gemm_bf16_kernel<1><<<dim3(D_MODEL / 128, ROWS / 128), 256, 0, stream>>>(
        attn, woT, nullptr, nullptr, out, b_out, ROWS, D_MODEL, D_MODEL);
}
